// CQT_70274254897349
// MI455X (gfx1250) — matrix operand order repaired and checked by reading, not compiled, not run on hardware
//
#include <hip/hip_runtime.h>
#include <hip/hip_bf16.h>
#include <cmath>

// ---- types for WMMA fragments ----
typedef __attribute__((ext_vector_type(16))) _Float16     v16h;
typedef __attribute__((ext_vector_type(8)))  float        v8f;
typedef __attribute__((ext_vector_type(4)))  unsigned int v4u;

#define CQT_T     220500   // samples per batch-channel
#define CQT_XSH   220504   // padded f16 row stride (16B-aligned rows)
#define CQT_HOP   512
#define CQT_NROW  192      // padded N (168 used: 84 bins x re/im)
#define CQT_KB    128      // K panel staged in LDS per round
#define CQT_LBS   136      // LDS row stride in halfs (128 + 8 pad, bank-spread)

// ---------------- 1) f32 -> f16 signal copy (padded stride) ----------------
__global__ void cqt_convert_kernel(const float* __restrict__ x,
                                   _Float16* __restrict__ xh) {
    int i = blockIdx.x * blockDim.x + threadIdx.x;
    if (i >= 16 * CQT_T) return;
    int bc = i / CQT_T;
    int l  = i - bc * CQT_T;
    xh[(size_t)bc * CQT_XSH + l] = (_Float16)x[i];
}

// ---------------- 2) windowed complex kernel bank (f16) --------------------
// row n: bin k = n>>1, n&1==0 -> re (win*cos), n&1==1 -> im (-win*sin)
__global__ void cqt_weights_kernel(_Float16* __restrict__ wB, int Kpad) {
    int idx = blockIdx.x * blockDim.x + threadIdx.x;
    int n = idx / Kpad;
    int l = idx - n * Kpad;
    if (n >= CQT_NROW) return;
    float val = 0.0f;
    if (n < 168) {
        int k = n >> 1;
        double freq = 32.7 * exp2((double)k / 12.0);
        double Q    = 1.0 / (exp2(1.0 / 12.0) - 1.0);
        int len     = (int)(Q * 22050.0 / freq + 0.5);
        if (l < len) {
            float t   = (float)l;
            float Lf  = (float)len;
            float win = 0.5f * (1.0f - cosf(6.28318530717958647f * t / Lf));
            float ph  = (float)(6.283185307179586 * freq / 22050.0) * t;
            float s, c;
            sincosf(ph, &s, &c);
            val = (n & 1) ? (-win * s) : (win * c);
        }
    }
    wB[(size_t)n * Kpad + l] = (_Float16)val;
}

// ---------------- 3) WMMA GEMM with LDS-shared B panel ---------------------
// 256 threads = 8 waves. Each wave: one 16-frame M tile; all waves share the
// block's 4 N-tiles (64 weight rows) staged through LDS in 128-wide K panels.
__global__ __launch_bounds__(256)
void cqt_wmma_kernel(const _Float16* __restrict__ xh,
                     const _Float16* __restrict__ wB,
                     float* __restrict__ out,
                     int nframes, int Kpad) {
    __shared__ __attribute__((aligned(16))) _Float16 lB[64 * CQT_LBS];

    const int tid  = threadIdx.x;
    const int lane = tid & 31;
    const int wave = tid >> 5;           // 0..7 -> M tile within block
    const int g    = lane >> 4;          // half-wave group
    const int mr   = lane & 15;
    const int ng   = blockIdx.x;         // group of 4 N tiles (0..2)
    const int mg   = blockIdx.y;         // group of 8 M tiles (0..3)
    const int bc   = blockIdx.z;         // batch*channel (0..15)

    const int n0 = ng * 64;              // first weight row for this block
    const int mt = mg * 8 + wave;        // this wave's M tile

    // A row for this lane: frame f (clamped; clamped rows never stored)
    int f = mt * 16 + mr;
    if (f >= nframes) f = nframes - 1;
    const _Float16* arow = xh + (size_t)bc * CQT_XSH + (size_t)f * CQT_HOP;

    const int a0  = 16 * g;   // A frag: K in [16g,16g+16)
    const int b0g = 16 * g;   // B frag: K in [16g,16g+16)

    v8f acc0 = {}, acc1 = {}, acc2 = {}, acc3 = {};

    for (int kk = 0; kk < Kpad; kk += CQT_KB) {
        __syncthreads();   // protect LDS panel against previous round's reads

        // Stage 64 rows x 128 halfs of B: 1024 16B granules, 4 per thread.
        // Consecutive threads fetch consecutive granules of a row (coalesced).
#pragma unroll
        for (int it = 0; it < 4; ++it) {
            int id  = it * 256 + tid;
            int row = id >> 4;           // 0..63
            int gc  = id & 15;           // granule within 128-half row
            *(v4u*)(lB + row * CQT_LBS + gc * 8) =
                *(const v4u*)(wB + (size_t)(n0 + row) * Kpad + kk + gc * 8);
        }
        __syncthreads();

#pragma unroll
        for (int s = 0; s < 4; ++s) {
            union { v16h h; v4u u[2]; } A;
            A.u[0] = *(const v4u*)(arow + kk + 32 * s + a0);
            A.u[1] = *(const v4u*)(arow + kk + 32 * s + a0 + 8);

            union { v16h h; v4u u[2]; } B0, B1, B2, B3;
            const _Float16* lb0 = lB + (0 * 16 + mr) * CQT_LBS + 32 * s + b0g;
            const _Float16* lb1 = lB + (1 * 16 + mr) * CQT_LBS + 32 * s + b0g;
            const _Float16* lb2 = lB + (2 * 16 + mr) * CQT_LBS + 32 * s + b0g;
            const _Float16* lb3 = lB + (3 * 16 + mr) * CQT_LBS + 32 * s + b0g;
            B0.u[0] = *(const v4u*)(lb0);  B0.u[1] = *(const v4u*)(lb0 + 8);
            B1.u[0] = *(const v4u*)(lb1);  B1.u[1] = *(const v4u*)(lb1 + 8);
            B2.u[0] = *(const v4u*)(lb2);  B2.u[1] = *(const v4u*)(lb2 + 8);
            B3.u[0] = *(const v4u*)(lb3);  B3.u[1] = *(const v4u*)(lb3 + 8);

            acc0 = __builtin_amdgcn_wmma_f32_16x16x32_f16(false, A.h, false, B0.h,
                                                          (short)0, acc0, false, false);
            acc1 = __builtin_amdgcn_wmma_f32_16x16x32_f16(false, A.h, false, B1.h,
                                                          (short)0, acc1, false, false);
            acc2 = __builtin_amdgcn_wmma_f32_16x16x32_f16(false, A.h, false, B2.h,
                                                          (short)0, acc2, false, false);
            acc3 = __builtin_amdgcn_wmma_f32_16x16x32_f16(false, A.h, false, B3.h,
                                                          (short)0, acc3, false, false);
        }
    }

    // D layout: lane -> N column (lane&15), VGPR j -> frame mt*16 + j + 8*g
    v8f accs[4] = {acc0, acc1, acc2, acc3};
#pragma unroll
    for (int nt = 0; nt < 4; ++nt) {
        int n = n0 + nt * 16 + mr;
        if (n < 168) {
            int k = n >> 1, reim = n & 1;
#pragma unroll
            for (int j = 0; j < 8; ++j) {
                int fr = mt * 16 + j + 8 * g;
                if (fr < nframes) {
                    out[(((size_t)bc * 84 + k) * nframes + fr) * 2 + reim] = accs[nt][j];
                }
            }
        }
    }
}

extern "C" void kernel_launch(void* const* d_in, const int* in_sizes, int n_in,
                              void* d_out, int out_size, void* d_ws, size_t ws_size,
                              hipStream_t stream) {
    (void)in_sizes; (void)n_in; (void)out_size; (void)ws_size;
    const float* x = (const float*)d_in[0];
    float* out = (float*)d_out;

    // Static CQT geometry (matches the reference's float64 setup)
    const double Q  = 1.0 / (std::pow(2.0, 1.0 / 12.0) - 1.0);
    const int Lmax  = (int)std::llround(Q * 22050.0 / 32.7);       // 11340
    const int nfr   = 1 + (CQT_T - Lmax) / CQT_HOP;                // 409
    const int Kpad  = ((Lmax + CQT_KB - 1) / CQT_KB) * CQT_KB;     // 11392

    // Workspace layout: [ f16 signal 16 x CQT_XSH | f16 weights 192 x Kpad ]
    _Float16* xh = (_Float16*)d_ws;
    _Float16* wB = xh + (size_t)16 * CQT_XSH;   // 16B-aligned offset

    {
        int total = 16 * CQT_T;
        cqt_convert_kernel<<<(total + 255) / 256, 256, 0, stream>>>(x, xh);
    }
    {
        int total = CQT_NROW * Kpad;
        cqt_weights_kernel<<<(total + 255) / 256, 256, 0, stream>>>(wB, Kpad);
    }
    {
        const int mtiles = (nfr + 15) / 16;          // 26
        dim3 grid(3, (mtiles + 7) / 8, 16);          // 3 x 4 x 16 blocks of 256
        cqt_wmma_kernel<<<grid, dim3(256), 0, stream>>>(xh, wB, out, nfr, Kpad);
    }
}
